// mLSTMCell_10831907520774
// MI455X (gfx1250) — matrix operand order repaired and checked by reading, not compiled, not run on hardware
//
#include <hip/hip_runtime.h>
#include <math.h>

// ---------------------------------------------------------------------------
// mLSTM parallel stabilized cell, flash-style, CDNA5 (gfx1250) WMMA bf16 path.
// B=2, S=2048, E=1024, NH=4, DH=256.
// ---------------------------------------------------------------------------

constexpr int B_ = 2, S_ = 2048, E_ = 1024, NH_ = 4, DH_ = 256;
constexpr float EPS_ = 1e-6f;
constexpr float QK_SCALE = 0.0625f;   // 1/sqrt(DH) = 1/16
constexpr float NEG_BIG = -1e30f;

typedef __attribute__((ext_vector_type(16))) __bf16 v16bf;
typedef __attribute__((ext_vector_type(8)))  float  v8f;

union BFrag {
  v16bf v;
  uint4 u[2];
  unsigned short s[16];
};

static __device__ __forceinline__ unsigned short f2bf(float f) {
  unsigned u = __float_as_uint(f);
  unsigned r = (u + 0x7FFFu + ((u >> 16) & 1u)) >> 16;   // RNE
  return (unsigned short)r;
}

// ---------------------------------------------------------------------------
// Kernel 0: fp32 -> bf16 head-split copies.
//   Qb, Kb: (B*NH, S, DH) row-major bf16
//   Vt:     (B*NH, DH, S) bf16 (transposed so C*V B-fragments are contiguous)
// ---------------------------------------------------------------------------
__global__ void prep_kernel(const float* __restrict__ q, const float* __restrict__ k,
                            const float* __restrict__ v, unsigned short* __restrict__ Qb,
                            unsigned short* __restrict__ Kb, unsigned short* __restrict__ Vt) {
  const size_t i = (size_t)blockIdx.x * 256 + threadIdx.x;
  if (i >= (size_t)B_ * S_ * E_) return;
  const int e = (int)(i % E_);
  const size_t bsI = i / E_;
  const int s = (int)(bsI % S_);
  const int b = (int)(bsI / S_);
  const int hh = e / DH_, d = e % DH_;
  const size_t hb = (size_t)(b * NH_ + hh);
  Qb[(hb * S_ + s) * DH_ + d] = f2bf(q[i]);
  Kb[(hb * S_ + s) * DH_ + d] = f2bf(k[i]);
  Vt[(hb * DH_ + d) * S_ + s] = f2bf(v[i]);
}

// ---------------------------------------------------------------------------
// Kernel 1: gate pre-activations. ig -> (B,NH,S); logfg = log_sigmoid(fg).
// One 128-thread block per (b,s).
// ---------------------------------------------------------------------------
__global__ void __launch_bounds__(128)
gates_kernel(const float* __restrict__ q, const float* __restrict__ k,
             const float* __restrict__ v,
             const float* __restrict__ igk, const float* __restrict__ igb,
             const float* __restrict__ fgk, const float* __restrict__ fgb,
             float* __restrict__ igv, float* __restrict__ logfg) {
  const int bs = blockIdx.x;
  const int b = bs / S_, s = bs % S_;
  const int tid = threadIdx.x;
  const float* qr = q + (size_t)bs * E_;
  const float* kr = k + (size_t)bs * E_;
  const float* vr = v + (size_t)bs * E_;
  float accI[NH_] = {0.f, 0.f, 0.f, 0.f};
  float accF[NH_] = {0.f, 0.f, 0.f, 0.f};
  for (int e = tid; e < E_; e += 128) {
    const float qe = qr[e], ke = kr[e], ve = vr[e];
#pragma unroll
    for (int hh = 0; hh < NH_; ++hh) {
      accI[hh] += qe * igk[e * NH_ + hh] + ke * igk[(E_ + e) * NH_ + hh] +
                  ve * igk[(2 * E_ + e) * NH_ + hh];
      accF[hh] += qe * fgk[e * NH_ + hh] + ke * fgk[(E_ + e) * NH_ + hh] +
                  ve * fgk[(2 * E_ + e) * NH_ + hh];
    }
  }
  __shared__ float red[128][8];
#pragma unroll
  for (int hh = 0; hh < NH_; ++hh) {
    red[tid][hh] = accI[hh];
    red[tid][NH_ + hh] = accF[hh];
  }
  __syncthreads();
  for (int st = 64; st > 0; st >>= 1) {
    if (tid < st) {
#pragma unroll
      for (int j = 0; j < 8; ++j) red[tid][j] += red[tid + st][j];
    }
    __syncthreads();
  }
  if (tid < NH_) {
    const int hh = tid;
    const size_t o = ((size_t)b * NH_ + hh) * S_ + s;
    igv[o] = red[0][hh] + igb[hh];
    const float fpre = red[0][NH_ + hh] + fgb[hh];
    // stable log_sigmoid
    logfg[o] = fminf(fpre, 0.f) - log1pf(__expf(-fabsf(fpre)));
  }
}

// ---------------------------------------------------------------------------
// Kernel 2: inclusive cumsum of logfg over S per (b,h). Hillis-Steele in LDS.
// ---------------------------------------------------------------------------
__global__ void __launch_bounds__(1024)
scan_kernel(const float* __restrict__ logfg, float* __restrict__ cg) {
  const int bh = blockIdx.x;
  __shared__ float buf[2][S_];
  const int t = threadIdx.x;
  buf[0][t] = logfg[(size_t)bh * S_ + t];
  buf[0][t + 1024] = logfg[(size_t)bh * S_ + t + 1024];
  __syncthreads();
  int src = 0;
  for (int off = 1; off < S_; off <<= 1) {
    const int dst = 1 - src;
    for (int i = t; i < S_; i += 1024) {
      float val = buf[src][i];
      if (i >= off) val += buf[src][i - off];
      buf[dst][i] = val;
    }
    __syncthreads();
    src = dst;
  }
  cg[(size_t)bh * S_ + t] = buf[src][t];
  cg[(size_t)bh * S_ + t + 1024] = buf[src][t + 1024];
}

// ---------------------------------------------------------------------------
// Kernel 3: flash-style mLSTM. Block = 128 threads (4 waves) handles one
// (b, h, 16-query tile). Wave w owns output dims [64w, 64w+64).
// Key tiles of 32. Online stabilization identical to reference semantics.
// Fused per-head layernorm epilogue.
// Software-pipelined: K/V fragments for tile jt+1 are loaded right after the
// WMMAs that consume tile jt's fragments, so load latency is hidden behind
// the stat phase / next matmul phase instead of stalling each WMMA.
// ---------------------------------------------------------------------------
__global__ void __launch_bounds__(128)
flash_mlstm(const unsigned short* __restrict__ Qb,
            const unsigned short* __restrict__ Kb,
            const unsigned short* __restrict__ Vt,
            const float* __restrict__ igv,
            const float* __restrict__ cg,
            const float* __restrict__ onw,
            float* __restrict__ out) {
  const int i0 = blockIdx.x * 16;
  const int bh = blockIdx.y;
  const int b = bh / NH_, h = bh % NH_;
  const int tid = threadIdx.x;
  const int wave = tid >> 5, lane = tid & 31;
  const int lr = lane & 15;         // row/col within 16
  const int hi = (lane >> 4) & 1;   // lane half
  const int rhalf = hi * 8;         // C/D frag: rows r+0 (lanes<16) / r+8 (lanes>=16)

  const unsigned short* Qh = Qb + (size_t)bh * S_ * DH_;
  const unsigned short* Kh = Kb + (size_t)bh * S_ * DH_;
  const unsigned short* Vh = Vt + (size_t)bh * DH_ * S_;
  const float* igh = igv + (size_t)bh * S_;
  const float* ch  = cg  + (size_t)bh * S_;

  __shared__ float sPart[4][16 * 32];        // per-wave partial QK^T tiles
  __shared__ unsigned short sCt[16 * 32];    // scaled C tile (bf16, row-major)
  __shared__ float sM[16], sL[16], sScale[16];
  __shared__ float sSum[4][16], sSq[4][16];

  if (tid < 16) { sM[tid] = NEG_BIG; sL[tid] = 0.f; }

  // Q A-fragments for this wave's two 32-wide K-chunks (ISA A layout:
  // lane holds row lr; halves map to dims {off..off+7} U {16+off..16+off+7}).
  BFrag aQ[2];
  {
    const unsigned short* qrow = Qh + (size_t)(i0 + lr) * DH_;
    const int off = hi * 8;
#pragma unroll
    for (int kc = 0; kc < 2; ++kc) {
      const int d0 = 64 * wave + 32 * kc;
      aQ[kc].u[0] = *reinterpret_cast<const uint4*>(qrow + d0 + off);
      aQ[kc].u[1] = *reinterpret_cast<const uint4*>(qrow + d0 + 16 + off);
    }
  }

  // Per-lane base pointers for K / V fragment loads (advance by tile).
  // K B-frag (QK^T): lane holds key row (j0 + 16n + lr), dims d0 + {hi*8..+7} U {16+hi*8..+7}
  const unsigned short* kbase = Kh + (size_t)lr * DH_ + 64 * wave + hi * 8;
  // V B-frag (C*V): lane holds dim row (64w + 16nt + lr), keys j0 + {hi*8..+7} U {16+hi*8..+7}
  const unsigned short* vbase = Vh + (size_t)(64 * wave + lr) * S_ + hi * 8;

  v8f acc[4];
#pragma unroll
  for (int nt = 0; nt < 4; ++nt)
#pragma unroll
    for (int r = 0; r < 8; ++r) acc[nt][r] = 0.f;

  // stat phase mapping: 8 threads per query row, 4 keys per thread
  const int srow = tid >> 3;
  const int scol = (tid & 7) * 4;
  const int gq = i0 + srow;
  const float cq = ch[gq];

  // ---- preload tile 0's K and V fragments (loop-carried) ----
  BFrag bk[2][2];   // [kc][n]
  BFrag bv[4];      // [nt]
#pragma unroll
  for (int kc = 0; kc < 2; ++kc)
#pragma unroll
    for (int n = 0; n < 2; ++n) {
      const unsigned short* krow = kbase + (size_t)(16 * n) * DH_ + 32 * kc;
      bk[kc][n].u[0] = *reinterpret_cast<const uint4*>(krow);
      bk[kc][n].u[1] = *reinterpret_cast<const uint4*>(krow + 16);
    }
#pragma unroll
  for (int nt = 0; nt < 4; ++nt) {
    const unsigned short* vrow = vbase + (size_t)(16 * nt) * S_;
    bv[nt].u[0] = *reinterpret_cast<const uint4*>(vrow);
    bv[nt].u[1] = *reinterpret_cast<const uint4*>(vrow + 16);
  }

  __syncthreads();

  const int nTiles = (i0 >> 5) + 1;   // causal: key tiles with j0 <= i0
  for (int jt = 0; jt < nTiles; ++jt) {
    const int j0 = jt * 32;
    const bool more = (jt + 1 < nTiles);

    // ---- QK^T partials over this wave's 64-dim slice: 4 WMMAs ----
    v8f ps[2];
#pragma unroll
    for (int n = 0; n < 2; ++n)
#pragma unroll
      for (int r = 0; r < 8; ++r) ps[n][r] = 0.f;
#pragma unroll
    for (int kc = 0; kc < 2; ++kc)
#pragma unroll
      for (int n = 0; n < 2; ++n)
        ps[n] = __builtin_amdgcn_wmma_f32_16x16x32_bf16(
            false, aQ[kc].v, false, bk[kc][n].v, (short)0, ps[n], false, false);

    // ---- pipeline: issue next tile's K-fragment loads now (covered by the
    //      stat phase below) ----
    if (more) {
      const int jn = j0 + 32;
#pragma unroll
      for (int kc = 0; kc < 2; ++kc)
#pragma unroll
        for (int n = 0; n < 2; ++n) {
          const unsigned short* krow = kbase + (size_t)(jn + 16 * n) * DH_ + 32 * kc;
          bk[kc][n].u[0] = *reinterpret_cast<const uint4*>(krow);
          bk[kc][n].u[1] = *reinterpret_cast<const uint4*>(krow + 16);
        }
    }
    // warm L2 two tiles ahead
    if (jt + 2 < nTiles) {
      const int jp = j0 + 64;
      __builtin_prefetch(kbase + (size_t)jp * DH_, 0, 1);
      __builtin_prefetch(vbase + jp, 0, 1);
    }

#pragma unroll
    for (int n = 0; n < 2; ++n)
#pragma unroll
      for (int r = 0; r < 8; ++r)
        sPart[wave][(r + rhalf) * 32 + 16 * n + lr] = ps[n][r];
    __syncthreads();

    // ---- decay matrix, masking, online max/sum (reference-exact math) ----
    float ld[4], qkv[4];
    bool msk[4];
    float mloc = NEG_BIG;
#pragma unroll
    for (int u = 0; u < 4; ++u) {
      const int col = scol + u, gk = j0 + col;
      const int idx = srow * 32 + col;
      qkv[u] = (sPart[0][idx] + sPart[1][idx] + sPart[2][idx] + sPart[3][idx]) * QK_SCALE;
      msk[u] = (gk <= gq);
      ld[u] = msk[u] ? (cq - ch[gk] + igh[gk]) : NEG_BIG;
      mloc = fmaxf(mloc, ld[u]);
    }
#pragma unroll
    for (int o = 4; o >= 1; o >>= 1) mloc = fmaxf(mloc, __shfl_xor(mloc, o, 32));
    const float mold = sM[srow];
    const float mnew = fmaxf(mold, mloc);
    const float rescale = __expf(mold - mnew);
    float ssum = 0.f;
#pragma unroll
    for (int u = 0; u < 4; ++u) {
      const float p = msk[u] ? __expf(ld[u] - mnew) * qkv[u] : 0.f;
      ssum += p;
      sCt[srow * 32 + scol + u] = f2bf(p);
    }
#pragma unroll
    for (int o = 4; o >= 1; o >>= 1) ssum += __shfl_xor(ssum, o, 32);
    if ((tid & 7) == 0) {
      sM[srow] = mnew;
      sL[srow] = sL[srow] * rescale + ssum;
      sScale[srow] = rescale;
    }
    __syncthreads();

    // ---- C x V: A(C) from LDS, V fragments were loaded one phase ahead ----
    BFrag aC;
    {
      const unsigned short* crow = sCt + lr * 32;
      const int off = hi * 8;
      aC.u[0] = *reinterpret_cast<const uint4*>(crow + off);
      aC.u[1] = *reinterpret_cast<const uint4*>(crow + 16 + off);
    }
#pragma unroll
    for (int r = 0; r < 8; ++r) {
      const float sc = sScale[r + rhalf];
#pragma unroll
      for (int nt = 0; nt < 4; ++nt) acc[nt][r] *= sc;
    }
#pragma unroll
    for (int nt = 0; nt < 4; ++nt)
      acc[nt] = __builtin_amdgcn_wmma_f32_16x16x32_bf16(
          false, aC.v, false, bv[nt].v, (short)0, acc[nt], false, false);

    // ---- pipeline: issue next tile's V-fragment loads (covered by the next
    //      QK^T + stat phases) ----
    if (more) {
      const int jn = j0 + 32;
#pragma unroll
      for (int nt = 0; nt < 4; ++nt) {
        const unsigned short* vrow = vbase + (size_t)(16 * nt) * S_ + jn;
        bv[nt].u[0] = *reinterpret_cast<const uint4*>(vrow);
        bv[nt].u[1] = *reinterpret_cast<const uint4*>(vrow + 16);
      }
    }
    // NOTE: no trailing barrier needed — sCt/sScale consumers above are
    // separated from next-iteration producers by the post-partial barrier.
  }

  // ---- epilogue: normalizer + fused multi-head layernorm ----
  float invn[8];
#pragma unroll
  for (int r = 0; r < 8; ++r) {
    const int row = r + rhalf;
    invn[r] = 1.0f / (fmaxf(fabsf(sL[row]), __expf(-sM[row])) + EPS_);
  }
#pragma unroll
  for (int r = 0; r < 8; ++r) {
    float s1 = 0.f, s2 = 0.f;
#pragma unroll
    for (int nt = 0; nt < 4; ++nt) {
      acc[nt][r] *= invn[r];
      s1 += acc[nt][r];
      s2 += acc[nt][r] * acc[nt][r];
    }
#pragma unroll
    for (int o = 1; o <= 8; o <<= 1) {
      s1 += __shfl_xor(s1, o, 32);
      s2 += __shfl_xor(s2, o, 32);
    }
    if (lr == 0) { sSum[wave][r + rhalf] = s1; sSq[wave][r + rhalf] = s2; }
  }
  __syncthreads();
#pragma unroll
  for (int r = 0; r < 8; ++r) {
    const int row = r + rhalf;
    const float s1 = sSum[0][row] + sSum[1][row] + sSum[2][row] + sSum[3][row];
    const float s2 = sSq[0][row] + sSq[1][row] + sSq[2][row] + sSq[3][row];
    const float mean = s1 * (1.0f / DH_);
    const float var = s2 * (1.0f / DH_) - mean * mean;
    const float rstd = rsqrtf(var + EPS_);
#pragma unroll
    for (int nt = 0; nt < 4; ++nt) {
      const int dim = 64 * wave + 16 * nt + lr;
      const float w = onw[h * DH_ + dim];
      out[((size_t)b * S_ + i0 + row) * E_ + h * DH_ + dim] =
          (acc[nt][r] - mean) * rstd * w;
    }
  }
}

// ---------------------------------------------------------------------------
extern "C" void kernel_launch(void* const* d_in, const int* in_sizes, int n_in,
                              void* d_out, int out_size, void* d_ws, size_t ws_size,
                              hipStream_t stream) {
  (void)in_sizes; (void)n_in; (void)out_size; (void)ws_size;
  const float* q   = (const float*)d_in[0];
  const float* k   = (const float*)d_in[1];
  const float* v   = (const float*)d_in[2];
  const float* igk = (const float*)d_in[3];
  const float* igb = (const float*)d_in[4];
  const float* fgk = (const float*)d_in[5];
  const float* fgb = (const float*)d_in[6];
  const float* onw = (const float*)d_in[7];
  float* out = (float*)d_out;

  // workspace layout: Qb | Kb | Vt (bf16, 8MB each) | ig | logfg | c (fp32)
  const size_t nElemH = (size_t)B_ * NH_ * S_ * DH_;   // 4M elements
  unsigned short* Qb = (unsigned short*)d_ws;
  unsigned short* Kb = Qb + nElemH;
  unsigned short* Vt = Kb + nElemH;
  float* igv   = (float*)(Vt + nElemH);
  float* logfg = igv + (size_t)B_ * NH_ * S_;
  float* cgv   = logfg + (size_t)B_ * NH_ * S_;

  prep_kernel<<<(B_ * S_ * E_ + 255) / 256, 256, 0, stream>>>(q, k, v, Qb, Kb, Vt);
  gates_kernel<<<B_ * S_, 128, 0, stream>>>(q, k, v, igk, igb, fgk, fgb, igv, logfg);
  scan_kernel<<<B_ * NH_, 1024, 0, stream>>>(logfg, cgv);
  flash_mlstm<<<dim3(S_ / 16, B_ * NH_), 128, 0, stream>>>(Qb, Kb, Vt, igv, cgv, onw, out);
}
